// Model_69724499083635
// MI455X (gfx1250) — compile-verified, matrix operand order checked by reading, not run on hardware
//
#include <hip/hip_runtime.h>
#include <math.h>

// ---------------- problem dims ----------------
#define NN    5000
#define EE    1250
#define NNZ_  160000
#define FIN   500
#define HIDD  512
#define KCH   10
#define NEGN  3500
#define N2    10000     // 2N rows for contrastive part

// padded dims (rows -> mult of 64 for 4x16 wave-tiling, K -> mult of 32)
#define FP    512       // K pad of F_IN
#define NP    5056      // NN padded to 64
#define N2P   10048     // 2N padded to 64
#define NEGP  7008      // 7000 padded to 32 (B-operand rows)
#define CATK  1024

typedef __attribute__((ext_vector_type(16))) __bf16 v16bf;
typedef __attribute__((ext_vector_type(8)))  float  v8f;

union FragBF { v16bf v; unsigned int u[8]; };

__device__ __forceinline__ unsigned short f2bf(float f) {
  unsigned int u = __float_as_uint(f);
  unsigned int r = (u + 0x7FFFu + ((u >> 16) & 1u)) >> 16;
  return (unsigned short)r;
}

__device__ __forceinline__ float wave_sum(float v) {
  #pragma unroll
  for (int s = 16; s >= 1; s >>= 1) v += __shfl_xor(v, s, 32);
  return v;
}

__device__ __forceinline__ float block_sum(float v, float* sm) {
  int t = threadIdx.x;
  sm[t] = v; __syncthreads();
  for (int s = blockDim.x >> 1; s > 0; s >>= 1) {
    if (t < s) sm[t] += sm[t + s];
    __syncthreads();
  }
  float r = sm[0]; __syncthreads();
  return r;
}

// ---------------- utility kernels ----------------
__global__ void hg_zero_f32(float* p, int n) {
  int i = blockIdx.x * 256 + threadIdx.x;
  if (i < n) p[i] = 0.0f;
}
__global__ void hg_zero_u16(unsigned short* p, int n) {
  int i = blockIdx.x * 256 + threadIdx.x;
  if (i < n) p[i] = 0;
}

// per-column mean / 1/(std+1e-6), ddof=1
__global__ void __launch_bounds__(256) hg_colstats(const float* __restrict__ x,
                                                   float* mean, float* rstd) {
  __shared__ float sm[256];
  int c = blockIdx.x;
  float s = 0.f, sq = 0.f;
  for (int r = threadIdx.x; r < NN; r += 256) {
    float v = x[(size_t)r * FIN + c];
    s += v; sq += v * v;
  }
  s  = block_sum(s, sm);
  sq = block_sum(sq, sm);
  if (threadIdx.x == 0) {
    float m = s / (float)NN;
    float var = (sq - (float)NN * m * m) / (float)(NN - 1);
    mean[c] = m;
    rstd[c] = 1.0f / (sqrtf(fmaxf(var, 0.f)) + 1e-6f);
  }
}

__global__ void hg_normalize(const float* __restrict__ x, const float* mean, const float* rstd,
                             float* of, unsigned short* ob) {
  int i = blockIdx.x * 256 + threadIdx.x;
  if (i >= NN * FIN) return;
  int r = i / FIN, c = i - r * FIN;
  float v = (x[i] - mean[c]) * rstd[c];
  of[i] = v;
  ob[(size_t)r * FP + c] = f2bf(v);
}

__global__ void hg_deg(const int* __restrict__ nidx, const int* __restrict__ eidx,
                       const float* __restrict__ w, float* dv, float* de) {
  int i = blockIdx.x * 256 + threadIdx.x;
  if (i >= NNZ_) return;
  atomicAdd(&dv[nidx[i]], w[i]);
  atomicAdd(&de[eidx[i]], w[i]);
}

__global__ void hg_invdeg(const float* dv, const float* de, float* dvis, float* dei) {
  int i = blockIdx.x * 256 + threadIdx.x;
  if (i < NN) dvis[i] = dv[i] > 0.f ? rsqrtf(fmaxf(dv[i], 1e-12f)) : 0.f;
  if (i < EE) dei[i]  = de[i] > 0.f ? 1.0f / fmaxf(de[i], 1e-12f) : 0.f;
}

// ---------------- sparse hypergraph operator ----------------
// m[edge] += (z[node] * dvis[node]) * w     (one block per incidence)
__global__ void __launch_bounds__(128) hg_scatter_ne(const float* __restrict__ z,
    const int* __restrict__ nidx, const int* __restrict__ eidx,
    const float* __restrict__ w, const float* __restrict__ dvis, float* m) {
  int i = blockIdx.x;
  int nd = nidx[i], ed = eidx[i];
  float s = w[i] * dvis[nd];
  const float* zr = z + (size_t)nd * FIN;
  float* mr = m + (size_t)ed * FIN;
  for (int f = threadIdx.x; f < FIN; f += 128)
    atomicAdd(&mr[f], zr[f] * s);
}

// tmp[node] += (m[edge] * dei[edge]) * w
__global__ void __launch_bounds__(128) hg_scatter_en(const float* __restrict__ m,
    const int* __restrict__ nidx, const int* __restrict__ eidx,
    const float* __restrict__ w, const float* __restrict__ dei, float* tmp) {
  int i = blockIdx.x;
  int nd = nidx[i], ed = eidx[i];
  float s = w[i] * dei[ed];
  const float* mr = m + (size_t)ed * FIN;
  float* tr = tmp + (size_t)nd * FIN;
  for (int f = threadIdx.x; f < FIN; f += 128)
    atomicAdd(&tr[f], mr[f] * s);
}

// T1 = -A(x);  acc = th0*x + th1*T1
__global__ void hg_cheb_first(const float* __restrict__ x, const float* __restrict__ tmp,
                              const float* __restrict__ dvis, const float* __restrict__ th,
                              float* t1, float* acc) {
  int i = blockIdx.x * 256 + threadIdx.x;
  if (i >= NN * FIN) return;
  int r = i / FIN;
  float v = -tmp[i] * dvis[r];
  t1[i] = v;
  acc[i] = th[0] * x[i] + th[1] * v;
}

// Tk = -2*A(T_{k-1}) - T_{k-2};  acc += th[k]*Tk
__global__ void hg_cheb_step(const float* __restrict__ tmp, const float* __restrict__ tprev2,
                             const float* __restrict__ dvis, const float* __restrict__ th,
                             int k, float* tk, float* acc) {
  int i = blockIdx.x * 256 + threadIdx.x;
  if (i >= NN * FIN) return;
  int r = i / FIN;
  float v = -2.0f * tmp[i] * dvis[r] - tprev2[i];
  tk[i] = v;
  acc[i] += th[k] * v;
}

__global__ void hg_tobf16(const float* __restrict__ a, unsigned short* ob) {
  int i = blockIdx.x * 256 + threadIdx.x;
  if (i >= NN * FIN) return;
  int r = i / FIN, c = i - r * FIN;
  ob[(size_t)r * FP + c] = f2bf(a[i]);
}

__global__ void hg_wconv(const float* __restrict__ w, int R, int C,
                         unsigned short* ob, int Cp) {
  int i = blockIdx.x * 256 + threadIdx.x;
  if (i >= R * C) return;
  int r = i / C, c = i - r * C;
  ob[(size_t)r * Cp + c] = f2bf(w[i]);
}

// ---------------- WMMA GEMM: C[M,Ncol] = act(A[.,K]bf16 @ B[Ncol,K]bf16^T + bias) ----------------
// act: 0 = none, 1 = relu, 2 = softshrink(1e-4)+relu  (== max(v-1e-4, 0))
__global__ void __launch_bounds__(256)
hg_wmma_gemm(const unsigned short* __restrict__ A, int lda,
             const unsigned short* __restrict__ B, int ldb,
             const float* __restrict__ bias,
             float* __restrict__ C, int ldc,
             unsigned short* __restrict__ Cb, int ldcb,
             int M, int Ncol, int K, int act) {
  const int lane = threadIdx.x & 31;
  const int wv   = threadIdx.x >> 5;
  const int tm = blockIdx.y * 64 + (wv & 3) * 16;
  const int tn = blockIdx.x * 32 + (wv >> 2) * 16;
  const int r16   = lane & 15;
  const int khalf = (lane >> 4) << 3;          // 0 or 8
  const unsigned short* arow = A + (size_t)(tm + r16) * lda;
  const unsigned short* brow = B + (size_t)(tn + r16) * ldb;
  v8f acc = {};
  for (int k0 = 0; k0 < K; k0 += 32) {
    FragBF fa, fb;
    #pragma unroll
    for (int p = 0; p < 4; ++p) {
      int ka = k0 + khalf + 2 * p;             // K block 0..15 half
      int kb = k0 + 16 + khalf + 2 * p;        // K block 16..31 half
      fa.u[p]     = *(const unsigned int*)(arow + ka);
      fa.u[p + 4] = *(const unsigned int*)(arow + kb);
      fb.u[p]     = *(const unsigned int*)(brow + ka);
      fb.u[p + 4] = *(const unsigned int*)(brow + kb);
    }
    acc = __builtin_amdgcn_wmma_f32_16x16x32_bf16(false, fa.v, false, fb.v,
                                                  (short)0, acc, false, false);
  }
  const int col   = tn + r16;
  const int mbase = tm + ((lane >> 4) << 3);
  float bcol = (bias && col < Ncol) ? bias[col] : 0.0f;
  #pragma unroll
  for (int r = 0; r < 8; ++r) {
    int m = mbase + r;
    if (m < M && col < Ncol) {
      float v = acc[r] + bcol;
      if (act == 1)      v = fmaxf(v, 0.0f);
      else if (act == 2) v = fmaxf(v - 1e-4f, 0.0f);
      if (C)  C[(size_t)m * ldc + col] = v;
      if (Cb) Cb[(size_t)m * ldcb + col] = f2bf(v);
    }
  }
}

// WMMA with exp()+row-sum epilogue for neg_sim exp sums
__global__ void __launch_bounds__(256)
hg_wmma_negexp(const unsigned short* __restrict__ A, int lda,
               const unsigned short* __restrict__ B, int ldb,
               const float* __restrict__ tempv, float* __restrict__ row_exp,
               int M, int Ncol, int K) {
  const int lane = threadIdx.x & 31;
  const int wv   = threadIdx.x >> 5;
  const int tm = blockIdx.y * 64 + (wv & 3) * 16;
  const int tn = blockIdx.x * 32 + (wv >> 2) * 16;
  const int r16   = lane & 15;
  const int khalf = (lane >> 4) << 3;
  const unsigned short* arow = A + (size_t)(tm + r16) * lda;
  const unsigned short* brow = B + (size_t)(tn + r16) * ldb;
  v8f acc = {};
  for (int k0 = 0; k0 < K; k0 += 32) {
    FragBF fa, fb;
    #pragma unroll
    for (int p = 0; p < 4; ++p) {
      int ka = k0 + khalf + 2 * p;
      int kb = k0 + 16 + khalf + 2 * p;
      fa.u[p]     = *(const unsigned int*)(arow + ka);
      fa.u[p + 4] = *(const unsigned int*)(arow + kb);
      fb.u[p]     = *(const unsigned int*)(brow + ka);
      fb.u[p + 4] = *(const unsigned int*)(brow + kb);
    }
    acc = __builtin_amdgcn_wmma_f32_16x16x32_bf16(false, fa.v, false, fb.v,
                                                  (short)0, acc, false, false);
  }
  const float invT = 1.0f / tempv[0];
  const int col   = tn + r16;
  const int mbase = tm + ((lane >> 4) << 3);
  #pragma unroll
  for (int r = 0; r < 8; ++r) {
    int m = mbase + r;
    float e = (m < M && col < Ncol) ? __expf(acc[r] * invT) : 0.0f;
    // reduce across the 16-lane half (cols) -> one atomic per row per wave
    #pragma unroll
    for (int s = 8; s >= 1; s >>= 1) e += __shfl_xor(e, s, 32);
    if ((lane & 15) == 0 && m < M) atomicAdd(&row_exp[m], e);
  }
}

// ---------------- row-wise kernels ----------------
__global__ void __launch_bounds__(256) hg_ln_relu(const float* __restrict__ x,
    const float* __restrict__ g, const float* __restrict__ b, unsigned short* ob) {
  __shared__ float sm[256];
  int row = blockIdx.x, t = threadIdx.x;
  size_t base = (size_t)row * HIDD;
  float v0 = x[base + t], v1 = x[base + t + 256];
  float mean = block_sum(v0 + v1, sm) / (float)HIDD;
  float d0 = v0 - mean, d1 = v1 - mean;
  float var = block_sum(d0 * d0 + d1 * d1, sm) / (float)HIDD;
  float rs = rsqrtf(var + 1e-5f);
  float y0 = fmaxf(d0 * rs * g[t] + b[t], 0.f);
  float y1 = fmaxf(d1 * rs * g[t + 256] + b[t + 256], 0.f);
  ob[(size_t)row * FP + t] = f2bf(y0);
  ob[(size_t)row * FP + t + 256] = f2bf(y1);
}

// h_cheb = l2n(relu(a*h1+b*h2)), h5n = l2n(h5), packed into catb[row, 0:1024] bf16
__global__ void __launch_bounds__(256) hg_cat(const float* __restrict__ h1,
    const float* __restrict__ h2, const float* __restrict__ h5,
    const float* alpha, const float* beta, unsigned short* catb) {
  __shared__ float sm[256];
  int row = blockIdx.x, t = threadIdx.x;
  size_t b = (size_t)row * HIDD;
  float al = alpha[0], be = beta[0];
  float a0 = fmaxf(al * h1[b + t] + be * h2[b + t], 0.f);
  float a1 = fmaxf(al * h1[b + t + 256] + be * h2[b + t + 256], 0.f);
  float f0 = h5[b + t], f1 = h5[b + t + 256];
  float na = block_sum(a0 * a0 + a1 * a1, sm);
  float nf = block_sum(f0 * f0 + f1 * f1, sm);
  float ia = 1.f / fmaxf(sqrtf(na), 1e-12f);
  float jf = 1.f / fmaxf(sqrtf(nf), 1e-12f);
  size_t cbo = (size_t)row * CATK;
  catb[cbo + t] = f2bf(a0 * ia);
  catb[cbo + t + 256] = f2bf(a1 * ia);
  catb[cbo + HIDD + t] = f2bf(f0 * jf);
  catb[cbo + HIDD + t + 256] = f2bf(f1 * jf);
}

// logits: [bil(h2,c), bil(h1,c), bil(h4,c), bil(h3,c)] with Wc = c @ bil_W^T precomputed
__global__ void __launch_bounds__(256) hg_logits(const float* __restrict__ h1,
    const float* __restrict__ h2, const float* __restrict__ h3, const float* __restrict__ h4,
    const float* __restrict__ Wc, const float* bilb, float* out) {
  int lane = threadIdx.x & 31, wv = threadIdx.x >> 5;
  int row = blockIdx.x * 8 + wv;
  if (row >= NN) return;
  size_t b = (size_t)row * HIDD;
  float d1 = 0, d2 = 0, d3 = 0, d4 = 0;
  for (int c = lane; c < HIDD; c += 32) {
    float wc = Wc[b + c];
    d1 += h2[b + c] * wc; d2 += h1[b + c] * wc;
    d3 += h4[b + c] * wc; d4 += h3[b + c] * wc;
  }
  d1 = wave_sum(d1); d2 = wave_sum(d2); d3 = wave_sum(d3); d4 = wave_sum(d4);
  if (lane == 0) {
    float bb = bilb[0];
    out[row] = d1 + bb; out[NN + row] = d2 + bb;
    out[2 * NN + row] = d3 + bb; out[3 * NN + row] = d4 + bb;
  }
}

// pos_sim + normalized h1/h2 rows (bf16) + column sums s1 (for neg mean trick)
__global__ void __launch_bounds__(256) hg_possim(const float* __restrict__ h1,
    const float* __restrict__ h2, const float* __restrict__ c,
    unsigned short* h12n, float* s1, float* pos_sim, float* pos_sum) {
  int lane = threadIdx.x & 31, wv = threadIdx.x >> 5;
  int row = blockIdx.x * 8 + wv;
  if (row >= NN) return;
  size_t b = (size_t)row * HIDD;
  float n1 = 0, n2 = 0, nc = 0, d1 = 0, d2 = 0;
  for (int k = lane; k < HIDD; k += 32) {
    float a = h1[b + k], bb = h2[b + k], cc = c[b + k];
    n1 += a * a; n2 += bb * bb; nc += cc * cc;
    d1 += a * cc; d2 += bb * cc;
  }
  n1 = wave_sum(n1); n2 = wave_sum(n2); nc = wave_sum(nc);
  d1 = wave_sum(d1); d2 = wave_sum(d2);
  float i1 = 1.f / fmaxf(sqrtf(n1), 1e-12f);
  float i2 = 1.f / fmaxf(sqrtf(n2), 1e-12f);
  float ic = 1.f / fmaxf(sqrtf(nc), 1e-12f);
  for (int k = lane; k < HIDD; k += 32) {
    float a = h1[b + k] * i1, bb = h2[b + k] * i2;
    h12n[(size_t)row * FP + k] = f2bf(a);
    h12n[(size_t)(NN + row) * FP + k] = f2bf(bb);
    atomicAdd(&s1[k], a + bb);
  }
  if (lane == 0) {
    float p1 = d1 * i1 * ic, p2 = d2 * i2 * ic;
    pos_sim[row] = p1; pos_sim[NN + row] = p2;
    atomicAdd(pos_sum, p1 + p2);
  }
}

// normalized negatives (bf16) + column sums s2
__global__ void __launch_bounds__(256) hg_negn(const float* __restrict__ h3,
    const float* __restrict__ h4, const int* __restrict__ neg_idx,
    unsigned short* negn, float* s2) {
  int lane = threadIdx.x & 31, wv = threadIdx.x >> 5;
  int row = blockIdx.x * 8 + wv;
  if (row >= 2 * NEGN) return;
  const float* src = (row < NEGN) ? (h3 + (size_t)neg_idx[row] * HIDD)
                                  : (h4 + (size_t)neg_idx[row - NEGN] * HIDD);
  float n = 0;
  for (int k = lane; k < HIDD; k += 32) { float v = src[k]; n += v * v; }
  n = wave_sum(n);
  float inv = 1.f / fmaxf(sqrtf(n), 1e-12f);
  for (int k = lane; k < HIDD; k += 32) {
    float v = src[k] * inv;
    negn[(size_t)row * FP + k] = f2bf(v);
    atomicAdd(&s2[k], v);
  }
}

// temp = 0.1 * clip(|mean(pos_sim) - mean(neg_sim)|, 0.5, 2)
// mean(neg_sim) = (colsum(h1n)+colsum(h2n)) . colsum(negn) / (2N * 2NEG)
__global__ void hg_temp(const float* s1, const float* s2, const float* pos_sum, float* tempv) {
  int lane = threadIdx.x;
  float d = 0;
  for (int k = lane; k < HIDD; k += 32) d += s1[k] * s2[k];
  d = wave_sum(d);
  if (lane == 0) {
    float negmean = d / ((float)N2 * (float)(2 * NEGN));
    float posmean = pos_sum[0] / (float)N2;
    float t = fabsf(posmean - negmean);
    t = fminf(fmaxf(t, 0.5f), 2.0f);
    tempv[0] = 0.1f * t;
  }
}

__global__ void hg_loss(const float* __restrict__ pos_sim, const float* __restrict__ row_exp,
                        const float* tempv, float* loss_acc) {
  int i = blockIdx.x * 256 + threadIdx.x;
  if (i >= N2) return;
  float t = tempv[0];
  float pe = __expf(pos_sim[i] / t);
  float nes = fminf(fmaxf(row_exp[i], 1e-10f), 1e10f);
  atomicAdd(loss_acc, logf(pe / (pe + nes)));
}

__global__ void hg_final(const float* loss_acc, float* out) {
  out[4 * NN] = -loss_acc[0] / (float)N2;
}

// ---------------- host orchestration ----------------
extern "C" void kernel_launch(void* const* d_in, const int* in_sizes, int n_in,
                              void* d_out, int out_size, void* d_ws, size_t ws_size,
                              hipStream_t stream) {
  (void)in_sizes; (void)n_in; (void)out_size; (void)ws_size;
  const int*   hyper  = (const int*)d_in[0];
  const int*   nidx   = hyper;
  const int*   eidx   = hyper + NNZ_;
  const float* feat   = (const float*)d_in[1];
  const float* shuf   = (const float*)d_in[2];
  const float* ew     = (const float*)d_in[3];
  const int*   negidx = (const int*)d_in[4];
  const float* th_h   = (const float*)d_in[5];
  const float* th_l   = (const float*)d_in[6];
  const float* lin1hW = (const float*)d_in[7];
  const float* lin1hb = (const float*)d_in[8];
  const float* lin1lW = (const float*)d_in[9];
  const float* lin1lb = (const float*)d_in[10];
  const float* lin2W  = (const float*)d_in[11];
  const float* lin2b  = (const float*)d_in[12];
  const float* ln1g   = (const float*)d_in[13];
  const float* ln1b   = (const float*)d_in[14];
  const float* ln2g   = (const float*)d_in[15];
  const float* ln2b   = (const float*)d_in[16];
  const float* ln5g   = (const float*)d_in[17];
  const float* ln5b   = (const float*)d_in[18];
  const float* ph1W   = (const float*)d_in[19];
  const float* ph1b   = (const float*)d_in[20];
  const float* ph2W   = (const float*)d_in[21];
  const float* ph2b   = (const float*)d_in[22];
  const float* ph5W   = (const float*)d_in[23];
  const float* ph5b   = (const float*)d_in[24];
  const float* pfW    = (const float*)d_in[25];
  const float* pfb    = (const float*)d_in[26];
  const float* alpha  = (const float*)d_in[27];
  const float* beta   = (const float*)d_in[28];
  const float* bilW   = (const float*)d_in[29];
  const float* bilb   = (const float*)d_in[30];
  float* out = (float*)d_out;

  // -------- workspace layout --------
  char* wsbase = (char*)d_ws;
  size_t off = 0;
  auto allocB = [&](size_t bytes) -> void* {
    size_t a = (off + 255) & ~(size_t)255;
    off = a + bytes;
    return (void*)(wsbase + a);
  };
  auto allocF = [&](size_t n) -> float* { return (float*)allocB(n * sizeof(float)); };
  auto allocH = [&](size_t n) -> unsigned short* { return (unsigned short*)allocB(n * 2); };

  float* featn = allocF((size_t)NN * FIN);
  float* shufn = allocF((size_t)NN * FIN);
  float* meanA = allocF(FIN); float* rstdA = allocF(FIN);
  float* meanB = allocF(FIN); float* rstdB = allocF(FIN);
  float* dv   = allocF(NN);  float* de  = allocF(EE);
  float* dvis = allocF(NN);  float* dei = allocF(EE);
  float* m_e  = allocF((size_t)EE * FIN);
  float* tmpn = allocF((size_t)NN * FIN);
  float* ta   = allocF((size_t)NN * FIN);
  float* tb   = allocF((size_t)NN * FIN);
  float* tc   = allocF((size_t)NN * FIN);
  float* accv = allocF((size_t)NN * FIN);
  float* hpre = allocF((size_t)NN * HIDD);
  float* h1 = allocF((size_t)NN * HIDD);
  float* h2 = allocF((size_t)NN * HIDD);
  float* h3 = allocF((size_t)NN * HIDD);
  float* h4 = allocF((size_t)NN * HIDD);
  float* h5 = allocF((size_t)NN * HIDD);
  float* cmat = allocF((size_t)NN * HIDD);
  float* Wc   = allocF((size_t)NN * HIDD);
  float* pos_sim  = allocF(N2);
  float* row_exp  = allocF(N2);
  float* s1 = allocF(HIDD);
  float* s2 = allocF(HIDD);
  float* pos_sum  = allocF(1);
  float* loss_acc = allocF(1);
  float* tempv    = allocF(1);

  // bf16 arena (contiguous so a single zero-fill clears all padding)
  size_t bfStart = (off + 255) & ~(size_t)255;
  unsigned short* featb  = allocH((size_t)NP * FP);
  unsigned short* shufb  = allocH((size_t)NP * FP);
  unsigned short* chb_hf = allocH((size_t)NP * FP);
  unsigned short* chb_lf = allocH((size_t)NP * FP);
  unsigned short* chb_hs = allocH((size_t)NP * FP);
  unsigned short* chb_ls = allocH((size_t)NP * FP);
  unsigned short* lnb    = allocH((size_t)NP * FP);
  unsigned short* catb   = allocH((size_t)NP * CATK);
  unsigned short* cb     = allocH((size_t)NP * FP);
  unsigned short* h12n   = allocH((size_t)N2P * FP);
  unsigned short* negnb  = allocH((size_t)NEGP * FP);
  unsigned short* w1hb = allocH((size_t)HIDD * FP);
  unsigned short* w1lb = allocH((size_t)HIDD * FP);
  unsigned short* w2b  = allocH((size_t)HIDD * FP);
  unsigned short* wp1b = allocH((size_t)HIDD * FP);
  unsigned short* wp2b = allocH((size_t)HIDD * FP);
  unsigned short* wp5b = allocH((size_t)HIDD * FP);
  unsigned short* wbilb = allocH((size_t)HIDD * FP);
  unsigned short* wpfb  = allocH((size_t)HIDD * CATK);
  size_t bfElems = (off - bfStart) / 2;

  auto zeroF = [&](float* p, int n) {
    hg_zero_f32<<<dim3((n + 255) / 256), 256, 0, stream>>>(p, n);
  };

  // -------- init accumulators / padded arenas --------
  hg_zero_u16<<<dim3((int)((bfElems + 255) / 256)), 256, 0, stream>>>(
      (unsigned short*)(wsbase + bfStart), (int)bfElems);
  zeroF(dv, NN); zeroF(de, EE);
  zeroF(s1, HIDD); zeroF(s2, HIDD);
  zeroF(pos_sum, 1); zeroF(loss_acc, 1);
  zeroF(row_exp, N2);

  // -------- standardize features --------
  hg_colstats<<<FIN, 256, 0, stream>>>(feat, meanA, rstdA);
  hg_colstats<<<FIN, 256, 0, stream>>>(shuf, meanB, rstdB);
  int nf_grid = (NN * FIN + 255) / 256;
  hg_normalize<<<nf_grid, 256, 0, stream>>>(feat, meanA, rstdA, featn, featb);
  hg_normalize<<<nf_grid, 256, 0, stream>>>(shuf, meanB, rstdB, shufn, shufb);

  // -------- degrees --------
  hg_deg<<<(NNZ_ + 255) / 256, 256, 0, stream>>>(nidx, eidx, ew, dv, de);
  hg_invdeg<<<(NN + 255) / 256, 256, 0, stream>>>(dv, de, dvis, dei);

  // -------- weight conversion (f32 -> padded bf16) --------
  auto wconv = [&](const float* w, int R, int C, unsigned short* ob, int Cp) {
    hg_wconv<<<(R * C + 255) / 256, 256, 0, stream>>>(w, R, C, ob, Cp);
  };
  wconv(lin1hW, HIDD, FIN, w1hb, FP);
  wconv(lin1lW, HIDD, FIN, w1lb, FP);
  wconv(lin2W,  HIDD, FIN, w2b,  FP);
  wconv(ph1W, HIDD, HIDD, wp1b, FP);
  wconv(ph2W, HIDD, HIDD, wp2b, FP);
  wconv(ph5W, HIDD, HIDD, wp5b, FP);
  wconv(bilW, HIDD, HIDD, wbilb, FP);
  wconv(pfW,  HIDD, CATK, wpfb, CATK);

  // -------- Chebyshev propagation (4 runs) --------
  auto applyA = [&](const float* zin) {
    zeroF(m_e, EE * FIN);
    hg_scatter_ne<<<NNZ_, 128, 0, stream>>>(zin, nidx, eidx, ew, dvis, m_e);
    zeroF(tmpn, NN * FIN);
    hg_scatter_en<<<NNZ_, 128, 0, stream>>>(m_e, nidx, eidx, ew, dei, tmpn);
  };
  auto chebprop = [&](const float* x, const float* theta, unsigned short* outb) {
    applyA(x);
    hg_cheb_first<<<nf_grid, 256, 0, stream>>>(x, tmpn, dvis, theta, ta, accv);
    const float* prev2 = x;
    const float* prev  = ta;
    float* bufs[3] = {ta, tb, tc};
    int nexti = 1;
    for (int k = 2; k <= KCH; ++k) {
      applyA(prev);
      float* cur = bufs[nexti];
      hg_cheb_step<<<nf_grid, 256, 0, stream>>>(tmpn, prev2, dvis, theta, k, cur, accv);
      prev2 = prev; prev = cur; nexti = (nexti + 1) % 3;
    }
    hg_tobf16<<<nf_grid, 256, 0, stream>>>(accv, outb);
  };
  chebprop(featn, th_h, chb_hf);
  chebprop(featn, th_l, chb_lf);
  chebprop(shufn, th_h, chb_hs);
  chebprop(shufn, th_l, chb_ls);

  // -------- WMMA GEMM helper --------
  auto gemm = [&](const unsigned short* A, int lda, const unsigned short* B, int ldb,
                  const float* bias, float* C, int ldc, unsigned short* Cb, int ldcb,
                  int M, int Mpad, int Ncol, int K, int act) {
    dim3 g(Ncol / 32, Mpad / 64);
    hg_wmma_gemm<<<g, 256, 0, stream>>>(A, lda, B, ldb, bias, C, ldc, Cb, ldcb,
                                        M, Ncol, K, act);
  };

  // encoder chain: h = (relu(LN(act(A@W1^T+b1))) @ Wp^T + bp)
  auto chain = [&](const unsigned short* Ab, const unsigned short* W1b, const float* b1,
                   const float* lg, const float* lb,
                   const unsigned short* Wpb, const float* bp, int act1, float* hout) {
    gemm(Ab, FP, W1b, FP, b1, hpre, HIDD, nullptr, 0, NN, NP, HIDD, FP, act1);
    hg_ln_relu<<<NN, 256, 0, stream>>>(hpre, lg, lb, lnb);
    gemm(lnb, FP, Wpb, FP, bp, hout, HIDD, nullptr, 0, NN, NP, HIDD, FP, 0);
  };
  chain(chb_hf, w1hb, lin1hb, ln1g, ln1b, wp1b, ph1b, 1, h1);
  chain(chb_lf, w1lb, lin1lb, ln2g, ln2b, wp2b, ph2b, 1, h2);
  chain(featb,  w2b,  lin2b,  ln5g, ln5b, wp5b, ph5b, 2, h5);  // softshrink+relu
  chain(chb_hs, w1hb, lin1hb, ln1g, ln1b, wp1b, ph1b, 1, h3);
  chain(chb_ls, w1lb, lin1lb, ln2g, ln2b, wp2b, ph2b, 1, h4);

  // -------- fuse + projection + bilinear --------
  hg_cat<<<NN, 256, 0, stream>>>(h1, h2, h5, alpha, beta, catb);
  gemm(catb, CATK, wpfb, CATK, pfb, cmat, HIDD, cb, FP, NN, NP, HIDD, CATK, 0);
  gemm(cb, FP, wbilb, FP, nullptr, Wc, HIDD, nullptr, 0, NN, NP, HIDD, FP, 0);
  hg_logits<<<(NN + 7) / 8, 256, 0, stream>>>(h1, h2, h3, h4, Wc, bilb, out);

  // -------- contrastive part --------
  hg_possim<<<(NN + 7) / 8, 256, 0, stream>>>(h1, h2, cmat, h12n, s1, pos_sim, pos_sum);
  hg_negn<<<(2 * NEGN + 7) / 8, 256, 0, stream>>>(h3, h4, negidx, negnb, s2);
  hg_temp<<<1, 32, 0, stream>>>(s1, s2, pos_sum, tempv);

  {
    dim3 g(NEGP / 32, N2P / 64);
    hg_wmma_negexp<<<g, 256, 0, stream>>>(h12n, FP, negnb, FP, tempv, row_exp,
                                          N2, 2 * NEGN, FP);
  }
  hg_loss<<<(N2 + 255) / 256, 256, 0, stream>>>(pos_sim, row_exp, tempv, loss_acc);
  hg_final<<<1, 1, 0, stream>>>(loss_acc, out);
}
